// MyGraphSage_3075196584645
// MI455X (gfx1250) — matrix operand order repaired and checked by reading, not compiled, not run on hardware
//
#include <hip/hip_runtime.h>

// ---------------------------------------------------------------------------
// GraphSAGE layer for MI455X (gfx1250, wave32, WMMA)
//   agg = scatter_mean(x[nbr] -> tgt);  out = l2norm(relu(x@Wl + agg@Wr + bl + br))
// N=100000, E=800000, D=OUT=128
// ---------------------------------------------------------------------------

typedef __attribute__((ext_vector_type(16))) __bf16 v16bf;
typedef __attribute__((ext_vector_type(8)))  float  v8f;
typedef __attribute__((ext_vector_type(4)))  int    v4i_t;

#define GS_D    128
#define GS_NT   8       // 128 output cols / 16
#define GS_KT   4       // 128 K / 32 per wmma
#define GS_FRAGS (2 * GS_KT * GS_NT)          // 64 B-fragments (Wl + Wr)
#define GS_WSWZ_ELEMS (GS_FRAGS * 32 * 16)    // 32768 bf16 = 64 KB

// Async global->LDS path (CDNA5 GLOBAL_LOAD_ASYNC_TO_LDS_B128, ASYNCcnt)
#if __has_builtin(__builtin_amdgcn_global_load_async_to_lds_b128)
#define GS_ASYNC 1
typedef __attribute__((address_space(1))) v4i_t* gv4i_p;   // global int4*
typedef __attribute__((address_space(3))) v4i_t* lv4i_p;   // LDS int4*
#else
#define GS_ASYNC 0
#endif

// ---- Kernel 1: zero the workspace (agg [N*128] + deg [N]) ------------------
__global__ void gs_zero(float* __restrict__ p, long long n) {
    long long i = (long long)blockIdx.x * blockDim.x + threadIdx.x;
    long long stride = (long long)gridDim.x * blockDim.x;
    for (; i < n; i += stride) p[i] = 0.0f;
}

// ---- Kernel 2: pre-swizzle weights into WMMA-B fragment-native bf16 --------
// wswz[((src*4+kt)*8+nt)*32*16 + lane*16 + i] = W_src[K][col]
//   K = kt*32 + i + (lane>>4)*16, col = nt*16 + (lane&15)
// so each lane's 16 B-elements are 32 contiguous bytes -> 2x ds_load_b128.
__global__ void gs_swizzle(const float* __restrict__ Wl,
                           const float* __restrict__ Wr,
                           __bf16* __restrict__ wswz) {
    int idx = blockIdx.x * blockDim.x + threadIdx.x;
    if (idx >= GS_WSWZ_ELEMS) return;
    int i    = idx & 15;
    int lane = (idx >> 4) & 31;
    int nt   = (idx >> 9) & 7;
    int kt   = (idx >> 12) & 3;
    int src  = idx >> 14;
    int K    = kt * 32 + i + (lane >> 4) * 16;
    int col  = nt * 16 + (lane & 15);
    const float* W = src ? Wr : Wl;
    wswz[idx] = (__bf16)W[K * GS_D + col];
}

// ---- Kernel 3: edge scatter-add (sum + degree) via L2 fp32 atomics ---------
// 32 lanes cooperate per edge; edge id forced wave-uniform so the index
// loads scalarize to SMEM and atomic addressing uses the SGPR base.
__global__ void gs_scatter(const float* __restrict__ x,
                           const int*   __restrict__ ei,
                           float* __restrict__ agg,
                           float* __restrict__ deg,
                           int E_) {
    long long tid  = (long long)blockIdx.x * blockDim.x + threadIdx.x;
    int       lane = (int)(tid & 31);
    long long e    = tid >> 5;
    if (e >= E_) return;
    int e_u = __builtin_amdgcn_readfirstlane((int)e);     // wave-uniform
    int tgt = ei[e_u];                                    // s_load
    int nbr = ei[E_ + e_u];                               // s_load
    const float4 v = ((const float4*)(x + (long long)nbr * GS_D))[lane];
    float* dst = agg + (long long)tgt * GS_D + lane * 4;
    atomicAdd(dst + 0, v.x);             // global_atomic_add_f32, non-returning
    atomicAdd(dst + 1, v.y);
    atomicAdd(dst + 2, v.z);
    atomicAdd(dst + 3, v.w);
    if (lane == 0) atomicAdd(deg + tgt, 1.0f);
}

// ---- Kernel 4: fused dual-GEMM + bias + relu + row L2 norm -----------------
// Block = 256 threads = 8 waves; each wave owns 16 node-rows x all 128 cols.
// 64KB fragment-ready weight image copied global->LDS (async if available).
__global__ __launch_bounds__(256)
void gs_gemm(const float* __restrict__ x,
             const float* __restrict__ bl,
             const float* __restrict__ br,
             const float* __restrict__ agg,
             const float* __restrict__ deg,
             const __bf16* __restrict__ wswz,
             float* __restrict__ out,
             int N_) {
    __shared__ v16bf sWf[GS_FRAGS * 32];   // 64 KB, fragment-native

    const int t = threadIdx.x;

    // Stage the pre-swizzled 64KB image into LDS, fully coalesced.
    {
        const uint4* s = (const uint4*)wswz;
        uint4*       d = (uint4*)sWf;
        const int nchunks = GS_WSWZ_ELEMS * 2 / 16;   // 4096 x 16B
#if GS_ASYNC
        for (int it = t; it < nchunks; it += 256) {
            __builtin_amdgcn_global_load_async_to_lds_b128(
                (gv4i_p)(s + it), (lv4i_p)(d + it), 0, 0);
        }
#if __has_builtin(__builtin_amdgcn_s_wait_asynccnt)
        __builtin_amdgcn_s_wait_asynccnt(0);
#else
        asm volatile("s_wait_asynccnt 0" ::: "memory");
#endif
#else
        for (int it = t; it < nchunks; it += 256) d[it] = s[it];
#endif
    }
    __syncthreads();

    const int wave = t >> 5;          // 0..7
    const int lane = t & 31;
    const int half = lane >> 4;       // 0: lanes 0-15, 1: lanes 16-31
    const int l16  = lane & 15;
    const int rowBase = blockIdx.x * 128 + wave * 16;

    // A-matrix row for this lane (clamped: WMMA needs EXEC all-ones; all
    // lanes compute, out-of-range rows are never stored).
    int arow = rowBase + l16;
    if (arow >= N_) arow = N_ - 1;
    const float invdeg = __builtin_amdgcn_rcpf(fmaxf(deg[arow], 1.0f));

    // Per-lane bias for each 16-wide N tile (C layout: col = nt*16 + l16).
    float bias[GS_NT];
#pragma unroll
    for (int nt = 0; nt < GS_NT; ++nt) {
        int col = nt * 16 + l16;
        bias[nt] = bl[col] + br[col];
    }

    v8f acc[GS_NT] = {};   // 64 VGPRs of f32 accumulators

    const float* xr = x   + (long long)arow * GS_D;
    const float* gr = agg + (long long)arow * GS_D;

    // Interleave the x@Wl and agg@Wr K-steps so both A-fragment load groups
    // are in flight together (both accumulate into the same acc).
#pragma unroll
    for (int kt = 0; kt < GS_KT; ++kt) {
        v16bf ax, ag;
#pragma unroll
        for (int i = 0; i < 16; ++i) {
            // A layout matches B image: half 0 holds K {0-15},
            // half 1 holds K {16-31}.
            int k = kt * 32 + i + half * 16;
            ax[i] = (__bf16)xr[k];
            ag[i] = (__bf16)(gr[k] * invdeg);   // fold 1/deg (row-scale)
        }
#pragma unroll
        for (int nt = 0; nt < GS_NT; ++nt) {
            v16bf b = sWf[(kt * GS_NT + nt) * 32 + lane];   // Wl frag
            acc[nt] = __builtin_amdgcn_wmma_f32_16x16x32_bf16(
                false, ax, false, b, (short)0, acc[nt], false, false);
        }
#pragma unroll
        for (int nt = 0; nt < GS_NT; ++nt) {
            v16bf b = sWf[((GS_KT + kt) * GS_NT + nt) * 32 + lane];   // Wr frag
            acc[nt] = __builtin_amdgcn_wmma_f32_16x16x32_bf16(
                false, ag, false, b, (short)0, acc[nt], false, false);
        }
    }

    // ---- epilogue: +bias, relu, row L2 norm, store ----
    // C/D layout: acc[nt][v] = row (rowBase + half*8 + v), col (nt*16 + l16).
    // A row lives across the 16 lanes of one half-wave -> butterfly over
    // masks 1,2,4,8 (never crosses the half boundary under wave32).
#pragma unroll
    for (int v = 0; v < 8; ++v) {
        int mrow = rowBase + half * 8 + v;
        float vals[GS_NT];
        float ss = 0.0f;
#pragma unroll
        for (int nt = 0; nt < GS_NT; ++nt) {
            float f = acc[nt][v] + bias[nt];
            f = fmaxf(f, 0.0f);
            vals[nt] = f;
            ss += f * f;
        }
        ss += __shfl_xor(ss, 1);
        ss += __shfl_xor(ss, 2);
        ss += __shfl_xor(ss, 4);
        ss += __shfl_xor(ss, 8);
        float invn = __builtin_amdgcn_rcpf(sqrtf(ss) + 1e-6f);
        if (mrow < N_) {
            float* orow = out + (long long)mrow * GS_D;
#pragma unroll
            for (int nt = 0; nt < GS_NT; ++nt)
                orow[nt * 16 + l16] = vals[nt] * invn;
        }
    }
}

// ---------------------------------------------------------------------------
extern "C" void kernel_launch(void* const* d_in, const int* in_sizes, int n_in,
                              void* d_out, int out_size, void* d_ws, size_t ws_size,
                              hipStream_t stream) {
    const float* x  = (const float*)d_in[0];
    const int*   ei = (const int*)  d_in[1];   // edge_index [2,E] (int32)
    const float* Wl = (const float*)d_in[2];
    const float* bl = (const float*)d_in[3];
    const float* Wr = (const float*)d_in[4];
    const float* br = (const float*)d_in[5];
    float*       out = (float*)d_out;

    const int N_ = in_sizes[0] / GS_D;   // 100000
    const int E_ = in_sizes[1] / 2;      // 800000

    float*  agg  = (float*)d_ws;                       // [N * 128]
    float*  deg  = agg + (size_t)N_ * GS_D;            // [N]
    __bf16* wswz = (__bf16*)(deg + N_);                // [32768] (64 KB), 16B-aligned

    // 1) zero agg + deg
    long long ztot = (long long)N_ * GS_D + N_;
    gs_zero<<<4096, 256, 0, stream>>>((float*)d_ws, ztot);

    // 2) pre-swizzle weights into fragment-native bf16 image (one-shot)
    gs_swizzle<<<(GS_WSWZ_ELEMS + 255) / 256, 256, 0, stream>>>(Wl, Wr, wswz);

    // 3) scatter-add over edges (32 lanes per edge)
    long long sthreads = (long long)E_ * 32;
    int sblocks = (int)((sthreads + 255) / 256);
    gs_scatter<<<sblocks, 256, 0, stream>>>(x, ei, agg, deg, E_);

    // 4) fused WMMA GEMM + epilogue (128 rows per block, 8 waves)
    int gblocks = (N_ + 127) / 128;
    gs_gemm<<<gblocks, 256, 0, stream>>>(x, bl, br, agg, deg, wswz, out, N_);
}
